// GR2ST_69415261438551
// MI455X (gfx1250) — matrix operand order repaired and checked by reading, not compiled, not run on hardware
//
#include <hip/hip_runtime.h>
#include <hip/hip_bf16.h>
#include <math.h>

#define NN   8192
#define IMGD 1024
#define DD   256
#define HPD  1024
#define EDG  (NN * 6)   // N*K_dyn + N self loops (K=5)

#define AS1 __attribute__((address_space(1)))
#define AS3 __attribute__((address_space(3)))

#if defined(__has_builtin)
#if __has_builtin(__builtin_amdgcn_global_load_async_to_lds_b128) && \
    __has_builtin(__builtin_amdgcn_s_wait_asynccnt)
#define HAVE_ASYNC_LDS 1
#endif
#endif
#ifndef HAVE_ASYNC_LDS
#define HAVE_ASYNC_LDS 0
#endif

typedef int    v4i   __attribute__((ext_vector_type(4)));
typedef float  v2f   __attribute__((ext_vector_type(2)));
typedef float  v8f   __attribute__((ext_vector_type(8)));
typedef __bf16 v16bf __attribute__((ext_vector_type(16)));

// ---------------------------------------------------------------------------
// CDNA5 fp32 WMMA: D(16x16,f32) = A(16x4,f32) x B(4x16,f32) + C
// ---------------------------------------------------------------------------
__device__ __forceinline__ v8f wmma_f32x4(v2f a, v2f b, v8f c) {
  return __builtin_amdgcn_wmma_f32_16x16x4_f32(false, a, false, b, (short)0, c,
                                               false, false);
}

// CDNA5 bf16 WMMA: D(16x16,f32) = A(16x32,bf16) x B(32x16,bf16) + C
__device__ __forceinline__ v8f wmma_bf16(v16bf a, v16bf b, v8f c) {
  return __builtin_amdgcn_wmma_f32_16x16x32_bf16(false, a, false, b, (short)0, c,
                                                 false, false);
}

union BFrag { unsigned short s[16]; v16bf v; };

// round-to-nearest-even f32 -> bf16
__device__ __forceinline__ unsigned short f2bf(float f) {
  unsigned u = __float_as_uint(f);
  u += 0x7FFFu + ((u >> 16) & 1u);
  return (unsigned short)(u >> 16);
}

// 16-byte global(bf16) -> LDS stage; async DMA (ASYNCcnt) when available.
__device__ __forceinline__ void stage16B(unsigned short* lds, const unsigned short* g) {
#if HAVE_ASYNC_LDS
  __builtin_amdgcn_global_load_async_to_lds_b128(
      (AS1 v4i*)const_cast<unsigned short*>(g), (AS3 v4i*)lds, 0, 0);
#else
  *reinterpret_cast<float4*>(lds) = *reinterpret_cast<const float4*>(g);
#endif
}
__device__ __forceinline__ void stage_wait() {
#if HAVE_ASYNC_LDS
  __builtin_amdgcn_s_wait_asynccnt(0);
#endif
}

__device__ __forceinline__ float gelu_exact(float x) {
  return 0.5f * x * (1.0f + erff(x * 0.70710678118654752f));
}

__device__ __forceinline__ float wave_sum(float s) {
#pragma unroll
  for (int off = 16; off; off >>= 1) s += __shfl_xor(s, off, 32);
  return s;
}

// monotone float<->uint mapping for atomicMax over signed floats
__device__ __forceinline__ unsigned fmap(float f) {
  unsigned b = __float_as_uint(f);
  return (b & 0x80000000u) ? ~b : (b | 0x80000000u);
}
__device__ __forceinline__ float funmap(unsigned u) {
  return (u & 0x80000000u) ? __uint_as_float(u ^ 0x80000000u)
                           : __uint_as_float(~u);
}

__device__ __forceinline__ void top5_insert(float s, int j, float tv[5], int ti[5]) {
  if (s > tv[4]) {
    int p = 4;
    while (p > 0 && s > tv[p - 1]) { tv[p] = tv[p - 1]; ti[p] = ti[p - 1]; --p; }
    tv[p] = s; ti[p] = j;
  }
}

// ---------------------------------------------------------------------------
// Generic fp32 WMMA GEMM: C[M,Nc] = act( preact(A)[M,Kd] @ W[Kd,Nc] + bias ) + resid
// block = 256 threads (8 waves), tile 64x64, BK=16.
// ---------------------------------------------------------------------------
__global__ __launch_bounds__(256)
void gr2st_gemm_kernel(const float* __restrict__ A, const float* __restrict__ W,
                       const float* __restrict__ bias, const float* __restrict__ resid,
                       float* __restrict__ C, int Kd, int Nc, int preGelu, int postRelu) {
  __shared__ float As[64][20];
  __shared__ float Bs[16][68];
  const int t = threadIdx.x, l = t & 31, w = t >> 5;
  const int wy = w >> 1, wx = w & 1;
  const int bm = blockIdx.y * 64, bn = blockIdx.x * 64;
  v8f c0 = {0, 0, 0, 0, 0, 0, 0, 0};
  v8f c1 = {0, 0, 0, 0, 0, 0, 0, 0};
  const int ar = t >> 2, ac = (t & 3) << 2;
  const int br = t >> 4, bc = (t & 15) << 2;
  for (int k0 = 0; k0 < Kd; k0 += 16) {
    float4 av = *reinterpret_cast<const float4*>(A + (size_t)(bm + ar) * Kd + k0 + ac);
    if (preGelu) {
      av.x = gelu_exact(av.x); av.y = gelu_exact(av.y);
      av.z = gelu_exact(av.z); av.w = gelu_exact(av.w);
    }
    As[ar][ac + 0] = av.x; As[ar][ac + 1] = av.y;
    As[ar][ac + 2] = av.z; As[ar][ac + 3] = av.w;
    float4 bv = *reinterpret_cast<const float4*>(W + (size_t)(k0 + br) * Nc + bn + bc);
    Bs[br][bc + 0] = bv.x; Bs[br][bc + 1] = bv.y;
    Bs[br][bc + 2] = bv.z; Bs[br][bc + 3] = bv.w;
    __syncthreads();
#pragma unroll
    for (int kk = 0; kk < 16; kk += 4) {
      const int ko = kk + ((l >> 4) << 1);
      const int m = l & 15;
      v2f a  = { As[wy * 16 + m][ko], As[wy * 16 + m][ko + 1] };
      v2f b0 = { Bs[ko][wx * 32 + m], Bs[ko + 1][wx * 32 + m] };
      v2f b1 = { Bs[ko][wx * 32 + 16 + m], Bs[ko + 1][wx * 32 + 16 + m] };
      c0 = wmma_f32x4(a, b0, c0);
      c1 = wmma_f32x4(a, b1, c1);
    }
    __syncthreads();
  }
  const int mbase = bm + wy * 16 + ((l >> 4) << 3);
  const int ncol = bn + wx * 32 + (l & 15);
#pragma unroll
  for (int v = 0; v < 8; ++v) {
    int m = mbase + v;
    float x0 = c0[v], x1 = c1[v];
    if (bias) { x0 += bias[ncol]; x1 += bias[ncol + 16]; }
    if (postRelu) { x0 = fmaxf(x0, 0.f); x1 = fmaxf(x1, 0.f); }
    if (resid) {
      x0 += resid[(size_t)m * Nc + ncol];
      x1 += resid[(size_t)m * Nc + ncol + 16];
    }
    C[(size_t)m * Nc + ncol] = x0;
    C[(size_t)m * Nc + ncol + 16] = x1;
  }
}

// ---------------------------------------------------------------------------
// LayerNorm over rows of length 256; one wave per row (wave32).
// ---------------------------------------------------------------------------
__global__ __launch_bounds__(256)
void gr2st_ln_kernel(const float* __restrict__ X, const float* __restrict__ g,
                     const float* __restrict__ b, float* __restrict__ Y) {
  const int w = threadIdx.x >> 5, l = threadIdx.x & 31;
  const int row = blockIdx.x * 8 + w;
  const float* x = X + (size_t)row * DD;
  float v[8]; float s = 0.f;
#pragma unroll
  for (int j = 0; j < 8; ++j) { v[j] = x[l + 32 * j]; s += v[j]; }
  s = wave_sum(s);
  const float mu = s * (1.f / 256.f);
  float q = 0.f;
#pragma unroll
  for (int j = 0; j < 8; ++j) { float d = v[j] - mu; q += d * d; }
  q = wave_sum(q);
  const float inv = rsqrtf(q * (1.f / 256.f) + 1e-5f);
  float* y = Y + (size_t)row * DD;
#pragma unroll
  for (int j = 0; j < 8; ++j) {
    int c = l + 32 * j;
    y[c] = (v[j] - mu) * inv * g[c] + b[c];
  }
}

// ---------------------------------------------------------------------------
// Elementwise helpers
// ---------------------------------------------------------------------------
__global__ void gr2st_fill_f32(float* p, float v, int n) {
  int i = blockIdx.x * 256 + threadIdx.x;
  if (i < n) p[i] = v;
}
__global__ void gr2st_fill_u32(unsigned* p, unsigned v, int n) {
  int i = blockIdx.x * 256 + threadIdx.x;
  if (i < n) p[i] = v;
}
__global__ void gr2st_f32_to_bf16(const float* __restrict__ x,
                                  unsigned short* __restrict__ y, int n) {
  int i = blockIdx.x * 256 + threadIdx.x;
  if (i < n) y[i] = f2bf(x[i]);
}
__global__ void gr2st_ctgather(const int* __restrict__ ct, const float* __restrict__ emb,
                               float* __restrict__ outp) {
  int i = blockIdx.x * 256 + threadIdx.x;
  if (i >= NN * 128) return;
  int n = i >> 7, c = i & 127;
  outp[i] = emb[ct[n] * 128 + c];
}
__global__ void gr2st_spot_combine(const float* __restrict__ expr, const float* __restrict__ pos,
                                   const float* __restrict__ xe, const float* __restrict__ ye,
                                   const float* __restrict__ ctln, float* __restrict__ spot) {
  int i = blockIdx.x * 256 + threadIdx.x;
  if (i >= NN * DD) return;
  int n = i >> 8, c = i & 255;
  int xi = (int)pos[2 * n + 0];
  int yi = (int)pos[2 * n + 1];
  spot[i] = expr[i] + xe[(size_t)xi * DD + c] + ye[(size_t)yi * DD + c] + ctln[i];
}
__global__ void gr2st_concat_bias(float* __restrict__ fin, const float* __restrict__ b1,
                                  const float* __restrict__ b2) {
  int i = blockIdx.x * 256 + threadIdx.x;
  if (i >= NN * 2048) return;
  int c = i & 2047;
  fin[i] += (c < 1024) ? b1[c] : b2[c - 1024];
}

// ---------------------------------------------------------------------------
// Dynamic graph top-5: scores = head @ tail^T + 0.3*same_ct (TEMP=1).
// bf16 WMMA 16x16x32, f32 accumulate; bf16 operands pre-converted in global.
// block = 256 threads handles 16 rows; streams 128-col x 128-K bf16 tiles
// via async global->LDS DMA. 8 waves each own one 16x16 tile.
// ---------------------------------------------------------------------------
__global__ __launch_bounds__(256)
void gr2st_dyn_topk(const unsigned short* __restrict__ head,
                    const unsigned short* __restrict__ tail,
                    const int* __restrict__ ct, int* __restrict__ outIdx) {
  __shared__ unsigned short Hs[16][264];   // full K=256 rows, bf16
  __shared__ unsigned short Ts[128][136];  // 128 cols x 128-K chunk, bf16
  __shared__ float Sc[16][132];
  __shared__ int cti[16];
  const int t = threadIdx.x, l = t & 31, w = t >> 5;
  const int bi = blockIdx.x * 16;
  for (int q = t; q < 512; q += 256) {                 // 16 rows x 32 chunks(8 bf16)
    int r = q >> 5, c8 = (q & 31) << 3;
    stage16B(&Hs[r][c8], head + (size_t)(bi + r) * DD + c8);
  }
  if (t < 16) cti[t] = ct[bi + t];
  float tv[5] = {-3e38f, -3e38f, -3e38f, -3e38f, -3e38f};
  int ti[5] = {0, 0, 0, 0, 0};
  stage_wait();
  __syncthreads();
  const int m = l & 15;
  const int abase = (l >> 4) << 3;    // +8 for upper half-lanes
  const int bbase = (l >> 4) << 4;    // +16 for upper half-lanes
  for (int j0 = 0; j0 < NN; j0 += 128) {
    v8f acc = {0, 0, 0, 0, 0, 0, 0, 0};
    for (int k0 = 0; k0 < DD; k0 += 128) {
      for (int q = t; q < 2048; q += 256) {            // 128 rows x 16 chunks(8 bf16)
        int r = q >> 4, c8 = (q & 15) << 3;
        stage16B(&Ts[r][c8], tail + (size_t)(j0 + r) * DD + k0 + c8);
      }
      stage_wait();
      __syncthreads();
#pragma unroll
      for (int kk = 0; kk < 128; kk += 32) {
        BFrag a, b;
#pragma unroll
        for (int e = 0; e < 8; ++e) {
          a.s[e]     = Hs[m][k0 + kk + abase + e];
          a.s[8 + e] = Hs[m][k0 + kk + abase + 16 + e];
        }
#pragma unroll
        for (int e = 0; e < 16; ++e) b.s[e] = Ts[w * 16 + m][kk + abase + (e & 7) + ((e >> 3) << 4)];
        acc = wmma_bf16(a.v, b.v, acc);
      }
      __syncthreads();
    }
#pragma unroll
    for (int v = 0; v < 8; ++v) {
      int mr = v + ((l >> 4) << 3);
      int n2 = w * 16 + m;
      Sc[mr][n2] = acc[v] + 0.3f * ((cti[mr] == ct[j0 + n2]) ? 1.f : 0.f);
    }
    __syncthreads();
    if (t < 16) {
      for (int n2 = 0; n2 < 128; ++n2) top5_insert(Sc[t][n2], j0 + n2, tv, ti);
    }
    __syncthreads();
  }
  if (t < 16) {
    for (int k = 0; k < 5; ++k) outIdx[(bi + t) * 5 + k] = ti[k];
  }
}

// ---------------------------------------------------------------------------
// Spatial 5-NN: top-5 of -sqrt(max(|pi|^2+|pj|^2-2 pi.pj,0)); one block/row.
// ---------------------------------------------------------------------------
__global__ __launch_bounds__(256)
void gr2st_spa_topk(const float* __restrict__ pos, int* __restrict__ outIdx) {
  __shared__ float cv[256 * 5];
  __shared__ int   ci[256 * 5];
  __shared__ float mv[32 * 5];
  __shared__ int   mi[32 * 5];
  const int i = blockIdx.x, t = threadIdx.x;
  const float xi = pos[2 * i], yi = pos[2 * i + 1];
  const float sqi = xi * xi + yi * yi;
  float tv[5] = {-3e38f, -3e38f, -3e38f, -3e38f, -3e38f};
  int ti[5] = {0, 0, 0, 0, 0};
  for (int j = t; j < NN; j += 256) {
    float xj = pos[2 * j], yj = pos[2 * j + 1];
    float d2 = sqi + xj * xj + yj * yj - 2.f * (xi * xj + yi * yj);
    top5_insert(-sqrtf(fmaxf(d2, 0.f)), j, tv, ti);
  }
  for (int k = 0; k < 5; ++k) { cv[t * 5 + k] = tv[k]; ci[t * 5 + k] = ti[k]; }
  __syncthreads();
  if (t < 32) {
    float v2_[5] = {-3e38f, -3e38f, -3e38f, -3e38f, -3e38f};
    int i2_[5] = {0, 0, 0, 0, 0};
    for (int q = t * 8; q < t * 8 + 8; ++q)
      for (int k = 0; k < 5; ++k) top5_insert(cv[q * 5 + k], ci[q * 5 + k], v2_, i2_);
    for (int k = 0; k < 5; ++k) { mv[t * 5 + k] = v2_[k]; mi[t * 5 + k] = i2_[k]; }
  }
  __syncthreads();
  if (t == 0) {
    float v3[5] = {-3e38f, -3e38f, -3e38f, -3e38f, -3e38f};
    int i3[5] = {0, 0, 0, 0, 0};
    for (int q = 0; q < 160; ++q) top5_insert(mv[q], mi[q], v3, i3);
    for (int k = 0; k < 5; ++k) outIdx[i * 5 + k] = i3[k];
  }
}

// ---------------------------------------------------------------------------
// GAT pieces
// ---------------------------------------------------------------------------
__global__ __launch_bounds__(256)
void gr2st_gat_alpha(const float* __restrict__ xw, const float* __restrict__ asrc,
                     const float* __restrict__ adst, float* __restrict__ aS,
                     float* __restrict__ aD) {
  const int n = blockIdx.x;
  const int l = threadIdx.x & 31, w = threadIdx.x >> 5;
  const int h = w & 3;
  const float* av = (w < 4) ? asrc : adst;
  const float* xr = xw + (size_t)n * HPD + h * 256;
  float s = 0.f;
  for (int c = l; c < 256; c += 32) s += xr[c] * av[h * 256 + c];
  s = wave_sum(s);
  if (l == 0) ((w < 4) ? aS : aD)[n * 4 + h] = s;
}

__global__ void gr2st_gat_edge_logit(const int* __restrict__ idx, const float* __restrict__ aS,
                                     const float* __restrict__ aD, float* __restrict__ elog,
                                     unsigned* __restrict__ emaxU) {
  int gidx = blockIdx.x * 256 + threadIdx.x;
  if (gidx >= EDG * 4) return;
  int h = gidx & 3, e = gidx >> 2;
  int n = e / 6, k = e - n * 6;
  int dst; bool masked = false;
  if (k < 5) { dst = idx[n * 5 + k]; masked = (dst == n); } else dst = n;
  float x = aS[n * 4 + h] + aD[dst * 4 + h];
  float lr = (x > 0.f) ? x : 0.2f * x;
  float lg = masked ? -1e9f : lr;
  elog[gidx] = lg;
  atomicMax(emaxU + (size_t)dst * 4 + h, fmap(lg));
}

__global__ void gr2st_gat_edge_exp(const int* __restrict__ idx, const float* __restrict__ elog,
                                   const unsigned* __restrict__ emaxU, float* __restrict__ ee,
                                   float* __restrict__ dsum) {
  int gidx = blockIdx.x * 256 + threadIdx.x;
  if (gidx >= EDG * 4) return;
  int h = gidx & 3, e = gidx >> 2;
  int n = e / 6, k = e - n * 6;
  int dst = (k < 5) ? idx[n * 5 + k] : n;
  float v = expf(elog[gidx] - funmap(emaxU[dst * 4 + h]));
  ee[gidx] = v;
  atomicAdd(dsum + (size_t)dst * 4 + h, v);
}

__global__ __launch_bounds__(256)
void gr2st_gat_scatter(const int* __restrict__ idx, const float* __restrict__ xw,
                       const float* __restrict__ ee, const float* __restrict__ dsum,
                       float* __restrict__ outp, int colOff) {
  const int e = blockIdx.x;
  const int n = e / 6, k = e - n * 6;
  const int src = n;
  const int dst = (k < 5) ? idx[n * 5 + k] : n;
  const int c = threadIdx.x;
#pragma unroll
  for (int h = 0; h < 4; ++h) {
    float wgt = ee[e * 4 + h] / dsum[dst * 4 + h];
    atomicAdd(outp + (size_t)dst * 2048 + colOff + h * 256 + c,
              wgt * xw[(size_t)src * HPD + h * 256 + c]);
  }
}

// ---------------------------------------------------------------------------
// Streaming row-logsumexp of X @ Y^T (TEMP=1) + diagonal capture.
// bf16 WMMA 16x16x32, f32 accumulate; same tiling as dyn_topk.
// ---------------------------------------------------------------------------
__global__ __launch_bounds__(256)
void gr2st_rowlse(const unsigned short* __restrict__ X,
                  const unsigned short* __restrict__ Y,
                  float* __restrict__ lse, float* __restrict__ diag) {
  __shared__ unsigned short Xs[16][264];
  __shared__ unsigned short Ts[128][136];
  __shared__ float Sc[16][132];
  const int t = threadIdx.x, l = t & 31, w = t >> 5;
  const int bi = blockIdx.x * 16;
  for (int q = t; q < 512; q += 256) {
    int r = q >> 5, c8 = (q & 31) << 3;
    stage16B(&Xs[r][c8], X + (size_t)(bi + r) * DD + c8);
  }
  float mx = -3e38f, sm = 0.f, dg = 0.f;
  stage_wait();
  __syncthreads();
  const int m = l & 15;
  const int abase = (l >> 4) << 3;
  const int bbase = (l >> 4) << 4;
  for (int j0 = 0; j0 < NN; j0 += 128) {
    v8f acc = {0, 0, 0, 0, 0, 0, 0, 0};
    for (int k0 = 0; k0 < DD; k0 += 128) {
      for (int q = t; q < 2048; q += 256) {
        int r = q >> 4, c8 = (q & 15) << 3;
        stage16B(&Ts[r][c8], Y + (size_t)(j0 + r) * DD + k0 + c8);
      }
      stage_wait();
      __syncthreads();
#pragma unroll
      for (int kk = 0; kk < 128; kk += 32) {
        BFrag a, b;
#pragma unroll
        for (int e = 0; e < 8; ++e) {
          a.s[e]     = Xs[m][k0 + kk + abase + e];
          a.s[8 + e] = Xs[m][k0 + kk + abase + 16 + e];
        }
#pragma unroll
        for (int e = 0; e < 16; ++e) b.s[e] = Ts[w * 16 + m][kk + abase + (e & 7) + ((e >> 3) << 4)];
        acc = wmma_bf16(a.v, b.v, acc);
      }
      __syncthreads();
    }
#pragma unroll
    for (int v = 0; v < 8; ++v) {
      int mr = v + ((l >> 4) << 3);
      Sc[mr][w * 16 + m] = acc[v];
    }
    __syncthreads();
    if (t < 16) {
      float tmax = -3e38f;
      for (int n2 = 0; n2 < 128; ++n2) tmax = fmaxf(tmax, Sc[t][n2]);
      float te = 0.f;
      for (int n2 = 0; n2 < 128; ++n2) te += expf(Sc[t][n2] - tmax);
      float nm = fmaxf(mx, tmax);
      sm = sm * expf(mx - nm) + te * expf(tmax - nm);
      mx = nm;
      int gi = bi + t;
      if (gi >= j0 && gi < j0 + 128) dg = Sc[t][gi - j0];
    }
    __syncthreads();
  }
  if (t < 16) {
    int gi = bi + t;
    lse[gi] = mx + logf(sm);
    diag[gi] = dg;
  }
}

__global__ __launch_bounds__(256)
void gr2st_loss(const float* __restrict__ lser, const float* __restrict__ lsec,
                const float* __restrict__ diag, float* __restrict__ out) {
  __shared__ float red[256];
  float s = 0.f;
  for (int i = threadIdx.x; i < NN; i += 256)
    s += (lser[i] - diag[i]) + (lsec[i] - diag[i]);
  red[threadIdx.x] = s;
  __syncthreads();
  for (int st = 128; st; st >>= 1) {
    if (threadIdx.x < st) red[threadIdx.x] += red[threadIdx.x + st];
    __syncthreads();
  }
  if (threadIdx.x == 0) out[0] = red[0] / (2.0f * (float)NN);
}

// ---------------------------------------------------------------------------
extern "C" void kernel_launch(void* const* d_in, const int* in_sizes, int n_in,
                              void* d_out, int out_size, void* d_ws, size_t ws_size,
                              hipStream_t stream) {
  (void)in_sizes; (void)n_in; (void)out_size; (void)ws_size;
  const float* image_features = (const float*)d_in[0];
  const float* expression     = (const float*)d_in[1];
  const float* position       = (const float*)d_in[2];
  const int*   cell_type      = (const int*)d_in[3];
  const float* ip_proj_w = (const float*)d_in[4];
  const float* ip_proj_b = (const float*)d_in[5];
  const float* ip_fc_w   = (const float*)d_in[6];
  const float* ip_fc_b   = (const float*)d_in[7];
  const float* ip_ln_g   = (const float*)d_in[8];
  const float* ip_ln_b   = (const float*)d_in[9];
  const float* xe        = (const float*)d_in[10];
  const float* ye        = (const float*)d_in[11];
  const float* ct_emb    = (const float*)d_in[12];
  const float* ct_lin_w  = (const float*)d_in[13];
  const float* ct_lin_b  = (const float*)d_in[14];
  const float* ct_ln_g   = (const float*)d_in[15];
  const float* ct_ln_b   = (const float*)d_in[16];
  const float* dh_w = (const float*)d_in[17];
  const float* dh_b = (const float*)d_in[18];
  const float* dt_w = (const float*)d_in[19];
  const float* dt_b = (const float*)d_in[20];
  const float* sp_w = (const float*)d_in[21];
  const float* sp_b = (const float*)d_in[22];
  const float* dg_w    = (const float*)d_in[23];
  const float* dg_asrc = (const float*)d_in[24];
  const float* dg_adst = (const float*)d_in[25];
  const float* dg_bias = (const float*)d_in[26];
  const float* sg_w    = (const float*)d_in[27];
  const float* sg_asrc = (const float*)d_in[28];
  const float* sg_adst = (const float*)d_in[29];
  const float* sg_bias = (const float*)d_in[30];
  const float* fu_w    = (const float*)d_in[31];
  const float* fu_b    = (const float*)d_in[32];
  const float* fu_ln_g = (const float*)d_in[33];
  const float* fu_ln_b = (const float*)d_in[34];
  const float* spp_proj_w = (const float*)d_in[35];
  const float* spp_proj_b = (const float*)d_in[36];
  const float* spp_fc_w   = (const float*)d_in[37];
  const float* spp_fc_b   = (const float*)d_in[38];
  const float* spp_ln_g   = (const float*)d_in[39];
  const float* spp_ln_b   = (const float*)d_in[40];

  float* ws = (float*)d_ws;
  size_t off = 0;
  auto alloc = [&](size_t n) { size_t r = off; off += n; return r; };
  const size_t o_imgproj = alloc((size_t)NN * DD);   // reused as spp_proj
  const size_t o_tmp     = alloc((size_t)NN * DD);   // pre-LN temp (both proj heads)
  const size_t o_imgemb  = alloc((size_t)NN * DD);
  const size_t o_ctg     = alloc((size_t)NN * 128);
  const size_t o_ctln    = alloc((size_t)NN * DD);
  const size_t o_spot    = alloc((size_t)NN * DD);   // reused as fused_relu
  const size_t o_head    = alloc((size_t)NN * DD);   // reused as spot_emb
  const size_t o_tail    = alloc((size_t)NN * DD);   // reused as fused_ln
  const size_t o_spin    = alloc((size_t)NN * DD);
  const size_t o_xw      = alloc((size_t)NN * HPD);
  const size_t o_fin     = alloc((size_t)NN * 2048);
  const size_t o_bfA     = alloc((size_t)NN * 128); // bf16 [N,256] operand A
  const size_t o_bfB     = alloc((size_t)NN * 128); // bf16 [N,256] operand B
  const size_t o_aS      = alloc((size_t)NN * 4);
  const size_t o_aD      = alloc((size_t)NN * 4);
  const size_t o_emax    = alloc((size_t)NN * 4);
  const size_t o_dsum    = alloc((size_t)NN * 4);
  const size_t o_elog    = alloc((size_t)EDG * 4);
  const size_t o_ee      = alloc((size_t)EDG * 4);
  const size_t o_dynidx  = alloc((size_t)NN * 5);
  const size_t o_spaidx  = alloc((size_t)NN * 5);
  const size_t o_lser    = alloc((size_t)NN);
  const size_t o_lsec    = alloc((size_t)NN);
  const size_t o_diag    = alloc((size_t)NN);

  float* imgproj = ws + o_imgproj;
  float* tmpb    = ws + o_tmp;
  float* imgemb  = ws + o_imgemb;
  float* ctg     = ws + o_ctg;
  float* ctln    = ws + o_ctln;
  float* spot    = ws + o_spot;
  float* headb   = ws + o_head;
  float* tailb   = ws + o_tail;
  float* spinb   = ws + o_spin;
  float* xw      = ws + o_xw;
  float* fin     = ws + o_fin;
  unsigned short* bfA = (unsigned short*)(ws + o_bfA);
  unsigned short* bfB = (unsigned short*)(ws + o_bfB);
  float* aS      = ws + o_aS;
  float* aD      = ws + o_aD;
  unsigned* emaxU = (unsigned*)(ws + o_emax);
  float* dsum    = ws + o_dsum;
  float* elog    = ws + o_elog;
  float* eebuf   = ws + o_ee;
  int* dynidx    = (int*)(ws + o_dynidx);
  int* spaidx    = (int*)(ws + o_spaidx);
  float* lser    = ws + o_lser;
  float* lsec    = ws + o_lsec;
  float* diagb   = ws + o_diag;

  const dim3 blk(256);
  const dim3 g256(DD / 64, NN / 64);
  const dim3 g1024(HPD / 64, NN / 64);
  const int gcv = (NN * DD + 255) / 256;

  // ---- image proj head ----
  gr2st_gemm_kernel<<<g256, blk, 0, stream>>>(image_features, ip_proj_w, ip_proj_b,
                                              nullptr, imgproj, IMGD, DD, 0, 0);
  gr2st_gemm_kernel<<<g256, blk, 0, stream>>>(imgproj, ip_fc_w, ip_fc_b,
                                              imgproj, tmpb, DD, DD, 1, 0);
  gr2st_ln_kernel<<<NN / 8, blk, 0, stream>>>(tmpb, ip_ln_g, ip_ln_b, imgemb);

  // ---- spot construction ----
  gr2st_ctgather<<<(NN * 128 + 255) / 256, blk, 0, stream>>>(cell_type, ct_emb, ctg);
  gr2st_gemm_kernel<<<g256, blk, 0, stream>>>(ctg, ct_lin_w, ct_lin_b,
                                              nullptr, tmpb, 128, DD, 0, 1);
  gr2st_ln_kernel<<<NN / 8, blk, 0, stream>>>(tmpb, ct_ln_g, ct_ln_b, ctln);
  gr2st_spot_combine<<<gcv, blk, 0, stream>>>(expression, position, xe, ye, ctln, spot);

  // ---- head / tail / spatial projections ----
  gr2st_gemm_kernel<<<g256, blk, 0, stream>>>(spot, dh_w, dh_b, nullptr, headb, DD, DD, 0, 0);
  gr2st_gemm_kernel<<<g256, blk, 0, stream>>>(spot, dt_w, dt_b, nullptr, tailb, DD, DD, 0, 0);
  gr2st_gemm_kernel<<<g256, blk, 0, stream>>>(spot, sp_w, sp_b, nullptr, spinb, DD, DD, 0, 0);

  // ---- graph construction (bf16 operands pre-converted in global) ----
  gr2st_f32_to_bf16<<<gcv, blk, 0, stream>>>(headb, bfA, NN * DD);
  gr2st_f32_to_bf16<<<gcv, blk, 0, stream>>>(tailb, bfB, NN * DD);
  gr2st_dyn_topk<<<NN / 16, blk, 0, stream>>>(bfA, bfB, cell_type, dynidx);
  gr2st_spa_topk<<<NN, blk, 0, stream>>>(position, spaidx);

  // ---- GATs scatter straight into concat buffer (stride 2048) ----
  gr2st_fill_f32<<<(NN * 2048 + 255) / 256, blk, 0, stream>>>(fin, 0.f, NN * 2048);

  const float* gx[2]    = {headb, spinb};
  const float* gw[2]    = {dg_w, sg_w};
  const float* gas[2]   = {dg_asrc, sg_asrc};
  const float* gad[2]   = {dg_adst, sg_adst};
  int* gidx[2]          = {dynidx, spaidx};
  const int gcol[2]     = {0, 1024};
  for (int g = 0; g < 2; ++g) {
    gr2st_gemm_kernel<<<g1024, blk, 0, stream>>>(gx[g], gw[g], nullptr, nullptr,
                                                 xw, DD, HPD, 0, 0);
    gr2st_gat_alpha<<<NN, blk, 0, stream>>>(xw, gas[g], gad[g], aS, aD);
    gr2st_fill_u32<<<(NN * 4 + 255) / 256, blk, 0, stream>>>(emaxU, 0x007FFFFFu, NN * 4);
    gr2st_fill_f32<<<(NN * 4 + 255) / 256, blk, 0, stream>>>(dsum, 0.f, NN * 4);
    gr2st_gat_edge_logit<<<(EDG * 4 + 255) / 256, blk, 0, stream>>>(gidx[g], aS, aD,
                                                                    elog, emaxU);
    gr2st_gat_edge_exp<<<(EDG * 4 + 255) / 256, blk, 0, stream>>>(gidx[g], elog, emaxU,
                                                                  eebuf, dsum);
    gr2st_gat_scatter<<<EDG, blk, 0, stream>>>(gidx[g], xw, eebuf, dsum, fin, gcol[g]);
  }
  gr2st_concat_bias<<<(NN * 2048 + 255) / 256, blk, 0, stream>>>(fin, dg_bias, sg_bias);

  // ---- fusion + spot proj head ----
  gr2st_gemm_kernel<<<g256, blk, 0, stream>>>(fin, fu_w, fu_b, nullptr, spot, 2048, DD, 0, 1);
  gr2st_ln_kernel<<<NN / 8, blk, 0, stream>>>(spot, fu_ln_g, fu_ln_b, tailb);
  gr2st_gemm_kernel<<<g256, blk, 0, stream>>>(tailb, spp_proj_w, spp_proj_b,
                                              nullptr, imgproj, DD, DD, 0, 0);
  gr2st_gemm_kernel<<<g256, blk, 0, stream>>>(imgproj, spp_fc_w, spp_fc_b,
                                              imgproj, tmpb, DD, DD, 1, 0);
  gr2st_ln_kernel<<<NN / 8, blk, 0, stream>>>(tmpb, spp_ln_g, spp_ln_b, headb); // spot_emb

  // ---- contrastive loss: streaming row/col logsumexp of cos ----
  gr2st_f32_to_bf16<<<gcv, blk, 0, stream>>>(headb, bfA, NN * DD);   // spot_emb
  gr2st_f32_to_bf16<<<gcv, blk, 0, stream>>>(imgemb, bfB, NN * DD);  // image_emb
  gr2st_rowlse<<<NN / 16, blk, 0, stream>>>(bfA, bfB, lser, diagb);
  gr2st_rowlse<<<NN / 16, blk, 0, stream>>>(bfB, bfA, lsec, diagb);
  gr2st_loss<<<1, blk, 0, stream>>>(lser, lsec, diagb, (float*)d_out);
}
